// MapEncoder_52355651338946
// MI455X (gfx1250) — compile-verified, matrix operand order checked by reading, not run on hardware
//
#include <hip/hip_runtime.h>
#include <hip/hip_bf16.h>

// ---------------------------------------------------------------------------
// MapEncoder forward for MI455X (gfx1250, wave32, WMMA).
// GEMMs: v_wmma_f32_16x16x32_f16; f32 activations in HBM, weights pre-packed
// to f16 [Npad][Kpad] (transposed) so all fragment loads are contiguous,
// unconditional b128 vector loads. Each wave computes a 16x(16*NT) tile,
// reusing its A fragment across NT WMMAs per K-step (NT=4 when Npad%64==0).
// ---------------------------------------------------------------------------

typedef __attribute__((ext_vector_type(16))) _Float16 v16h;
typedef __attribute__((ext_vector_type(8)))  _Float16 v8h;
typedef __attribute__((ext_vector_type(8)))  float    v8f;

#define BB      4
#define NTOK    768
#define PPTS    20
#define HH      256
#define PE_D    128
#define NHEAD   8
#define DHEAD   32
#define DFFD    1024
#define DRPED   64
#define LLAY    6
#define KNNK    32
#define ATTRD   32

// ---------------- weight pack: W[K][N] f32 -> Wh[Npad][Kpad] f16 (transposed,
// zero-padded) so B fragments are contiguous in K ---------------------------
__global__ void pack_w_kernel(const float* __restrict__ W, int ldw, int K, int N,
                              _Float16* __restrict__ Wh, int Kpad, int total)
{
  int i = blockIdx.x * 256 + threadIdx.x;
  if (i >= total) return;
  int n = i / Kpad;
  int k = i % Kpad;
  float v = (k < K && n < N) ? W[(size_t)k * ldw + n] : 0.f;
  Wh[i] = (_Float16)v;
}

// ---------------- WMMA GEMM: C = act(A[M,Kpad]_f32 @ Wh^T + bias) ----------
// Requirements: M%16==0, Kpad%32==0, Npad%(16*NT)==0.
template <int NT>
__global__ __launch_bounds__(256)
void gemm_wmma_f16(const float* __restrict__ A, int lda,
                   const _Float16* __restrict__ Wh,   // [Npad][Kpad]
                   const float* __restrict__ bias, int biasN,
                   float* __restrict__ C, int ldc,
                   int M, int Kpad, int Npad, int act)
{
  int wave = blockIdx.x * 8 + (threadIdx.x >> 5);
  int ntn  = Npad / (16 * NT);
  int tm   = wave / ntn;
  int tn   = wave % ntn;
  if (tm * 16 >= M) return;                 // wave-uniform exit
  int lane = threadIdx.x & 31;
  int half = lane >> 4;                     // lo: K 0..7,16..23 ; hi: K 8..15,24..31
  int lr   = lane & 15;
  int arow = tm * 16 + lr;
  const float* Arow = A + (size_t)arow * lda + half * 8;
  const _Float16* Wrow[NT];
#pragma unroll
  for (int nt = 0; nt < NT; ++nt)
    Wrow[nt] = Wh + (size_t)(tn * NT * 16 + nt * 16 + lr) * Kpad + half * 8;

  v8f acc[NT];
#pragma unroll
  for (int nt = 0; nt < NT; ++nt)
    acc[nt] = (v8f){0.f, 0.f, 0.f, 0.f, 0.f, 0.f, 0.f, 0.f};

  for (int k0 = 0; k0 < Kpad; k0 += 32) {
    __builtin_prefetch(Arow + k0 + 32, 0, 1);
    v8f afl = *(const v8f*)(Arow + k0);        // K = k0+half*8 .. +7
    v8f afh = *(const v8f*)(Arow + k0 + 16);   // K = k0+16+half*8 .. +7
    v8h alo = __builtin_convertvector(afl, v8h);
    v8h ahi = __builtin_convertvector(afh, v8h);
    v16h a  = __builtin_shufflevector(alo, ahi, 0,1,2,3,4,5,6,7,8,9,10,11,12,13,14,15);
#pragma unroll
    for (int nt = 0; nt < NT; ++nt) {
      v8h b0 = *(const v8h*)(Wrow[nt] + k0);
      v8h b1 = *(const v8h*)(Wrow[nt] + k0 + 16);
      v16h b = __builtin_shufflevector(b0, b1, 0,1,2,3,4,5,6,7,8,9,10,11,12,13,14,15);
      acc[nt] = __builtin_amdgcn_wmma_f32_16x16x32_f16(
          false, a, false, b, (short)0, acc[nt], false, false);
    }
  }

#pragma unroll
  for (int nt = 0; nt < NT; ++nt) {
    int bcol = tn * NT * 16 + nt * 16 + lr;
    float bb = (bias && bcol < biasN) ? bias[bcol] : 0.f;
#pragma unroll
    for (int j = 0; j < 8; ++j) {
      int row = tm * 16 + j + half * 8;      // C/D layout: VGPR j -> M=j (+8 hi)
      float v = acc[nt][j] + bb;
      if (act) v = fmaxf(v, 0.f);
      C[(size_t)row * ldc + bcol] = v;
    }
  }
}

// ---------------- per-point local-frame features (pe_in, stride 32) --------
__global__ void prep_points_kernel(const float* __restrict__ pose,
                                   float* __restrict__ pe_in, int total)
{
  int idx = blockIdx.x * 256 + threadIdx.x;
  if (idx >= total) return;
  int bn = idx / PPTS;
  const float* tp = pose + (size_t)bn * PPTS * 3;   // point 0 = token pose
  const float* pp = pose + (size_t)idx * 3;
  float cy = cosf(tp[2]), sy = sinf(tp[2]);
  float dx = pp[0] - tp[0], dy = pp[1] - tp[1];
  float dyaw = pp[2] - tp[2];
  float* o = pe_in + (size_t)idx * 32;
  o[0] =  cy * dx + sy * dy;
  o[1] = -sy * dx + cy * dy;
  o[2] = cosf(dyaw);
  o[3] = sinf(dyaw);
#pragma unroll
  for (int z = 4; z < 32; ++z) o[z] = 0.f;          // K padding
}

// ---------------- one-hot part of attr MLP, fused with relu ----------------
__global__ void ha_kernel(const float* __restrict__ abase,
                          const float* __restrict__ in_wa,
                          const float* __restrict__ in_ba,
                          float* __restrict__ h, int total)
{
  int i = blockIdx.x * 256 + threadIdx.x;
  if (i >= total) return;
  int row = i >> 7;
  int c   = i & 127;
  int bn  = row / PPTS;
  int p   = row % PPTS;
  float v = abase[(size_t)bn * 128 + c] + in_wa[(size_t)(32 + p) * 128 + c] + in_ba[c];
  h[(size_t)row * HH + c] = fmaxf(v, 0.f);
}

// ---------------- masked max-pool over P points ----------------------------
__global__ void pool_kernel(const float* __restrict__ hp,
                            const unsigned char* __restrict__ valid,
                            float* __restrict__ pooled)
{
  int bn = blockIdx.x;
  int c  = threadIdx.x;
  float m = -1e9f;
  for (int p = 0; p < PPTS; ++p) {
    unsigned char vv = valid[(size_t)bn * PPTS + p];
    float val = vv ? hp[((size_t)bn * PPTS + p) * HH + c] : -1e9f;
    m = fmaxf(m, val);
  }
  unsigned char qv = valid[(size_t)bn * PPTS];
  pooled[(size_t)bn * HH + c] = qv ? m : 0.f;
}

// ---------------- KNN selection + relative-pose features -------------------
__global__ void knn_kernel(const float* __restrict__ pose,
                           const unsigned char* __restrict__ valid,
                           int* __restrict__ knn_idx, int* __restrict__ knn_inv,
                           float* __restrict__ rpe_in)
{
  int bn = blockIdx.x;
  int b  = bn / NTOK;
  int i  = bn % NTOK;
  int t  = threadIdx.x;
  __shared__ float dist[NTOK];
  __shared__ float rv[256];
  __shared__ int   ri[256];
  __shared__ int   selj[KNNK];
  __shared__ float selv[KNNK];

  const float* qp = pose + ((size_t)(b * NTOK + i) * PPTS) * 3;
  float xi = qp[0], yi = qp[1], yawi = qp[2];

  for (int j = t; j < NTOK; j += 256) {
    const float* kp = pose + ((size_t)(b * NTOK + j) * PPTS) * 3;
    float dx = kp[0] - xi, dy = kp[1] - yi;
    float dd = sqrtf(dx * dx + dy * dy + 1e-12f);
    unsigned char kv = valid[(size_t)(b * NTOK + j) * PPTS];
    dist[j] = kv ? dd : 1e6f;
  }
  __syncthreads();

  for (int ks = 0; ks < KNNK; ++ks) {
    float bestv = 3.4e38f;
    int   bestj = NTOK;
    for (int j = t; j < NTOK; j += 256) {
      float v = dist[j];
      if (v < bestv) { bestv = v; bestj = j; }
    }
    rv[t] = bestv; ri[t] = bestj;
    __syncthreads();
    for (int off = 128; off; off >>= 1) {
      if (t < off) {
        float v2 = rv[t + off]; int j2 = ri[t + off];
        if (v2 < rv[t] || (v2 == rv[t] && j2 < ri[t])) { rv[t] = v2; ri[t] = j2; }
      }
      __syncthreads();
    }
    if (t == 0) { selj[ks] = ri[0]; selv[ks] = rv[0]; dist[ri[0]] = 3.0e38f; }
    __syncthreads();
  }

  unsigned char qv = valid[(size_t)(b * NTOK + i) * PPTS];
  if (t < KNNK) {
    int j = selj[t];
    knn_idx[(size_t)bn * KNNK + t] = j;
    knn_inv[(size_t)bn * KNNK + t] = ((selv[t] > 1500.f) || (!qv)) ? 1 : 0;
    const float* kp = pose + ((size_t)(b * NTOK + j) * PPTS) * 3;
    float dx = kp[0] - xi, dy = kp[1] - yi;
    float ci = cosf(yawi), si = sinf(yawi);
    float dyaw = kp[2] - yawi;
    float* o = rpe_in + ((size_t)bn * KNNK + t) * 32;
    o[0] =  ci * dx + si * dy;
    o[1] = -si * dx + ci * dy;
    o[2] = cosf(dyaw);
    o[3] = sinf(dyaw);
#pragma unroll
    for (int z = 4; z < 32; ++z) o[z] = 0.f;        // K padding
  }
}

// ---------------- attention: one block per token, one wave per head --------
__global__ __launch_bounds__(256)
void attn_kernel(const float* __restrict__ q, const float* __restrict__ kk,
                 const float* __restrict__ vv, const float* __restrict__ bias,
                 const int* __restrict__ knn, const int* __restrict__ knninv,
                 float* __restrict__ o)
{
  int bn = blockIdx.x;
  int b  = bn / NTOK;
  int t  = threadIdx.x;
  int h  = t >> 5;
  int lane = t & 31;
  __shared__ float qs[HH];
  __shared__ int   idx[KNNK];
  __shared__ float inv[KNNK];

  qs[t] = q[(size_t)bn * HH + t];
  if (t < KNNK) {
    idx[t] = b * NTOK + knn[(size_t)bn * KNNK + t];
    inv[t] = knninv[(size_t)bn * KNNK + t] ? 1.f : 0.f;
  }
  __syncthreads();

  int krow = idx[lane];
  float lg = 0.f;
  for (int d = 0; d < DHEAD; ++d)
    lg += qs[h * DHEAD + d] * kk[(size_t)krow * HH + h * DHEAD + d];
  lg = lg * 0.17677669529663687f + bias[((size_t)bn * KNNK + lane) * 16 + h];
  if (inv[lane] != 0.f) lg = -1e9f;

  float m = lg;
  for (int off = 16; off; off >>= 1) m = fmaxf(m, __shfl_xor(m, off, 32));
  float e = expf(lg - m);
  float s = e;
  for (int off = 16; off; off >>= 1) s += __shfl_xor(s, off, 32);
  float a = e / s;

  float acc = 0.f;
  for (int k2 = 0; k2 < KNNK; ++k2) {
    float ak = __shfl(a, k2, 32);
    acc += ak * vv[(size_t)idx[k2] * HH + h * DHEAD + lane];
  }
  o[(size_t)bn * HH + h * DHEAD + lane] = acc;
}

// ---------------- fused residual add + LayerNorm (in-place on x) -----------
__global__ void add_ln_kernel(float* __restrict__ x, const float* __restrict__ r,
                              const float* __restrict__ g, const float* __restrict__ b)
{
  int row = blockIdx.x;
  int t   = threadIdx.x;
  __shared__ float sh[HH];
  float v = x[(size_t)row * HH + t] + r[(size_t)row * HH + t];
  sh[t] = v;
  __syncthreads();
  for (int off = 128; off; off >>= 1) { if (t < off) sh[t] += sh[t + off]; __syncthreads(); }
  float mean = sh[0] / (float)HH;
  __syncthreads();
  float d = v - mean;
  sh[t] = d * d;
  __syncthreads();
  for (int off = 128; off; off >>= 1) { if (t < off) sh[t] += sh[t + off]; __syncthreads(); }
  float var = sh[0] / (float)HH;
  x[(size_t)row * HH + t] = d * rsqrtf(var + 1e-5f) * g[t] + b[t];
}

// ---------------- final masking --------------------------------------------
__global__ void final_kernel(const float* __restrict__ x,
                             const unsigned char* __restrict__ valid,
                             float* __restrict__ out)
{
  int i = blockIdx.x * 256 + threadIdx.x;
  int bn = i >> 8;
  out[i] = valid[(size_t)bn * PPTS] ? x[i] : 0.f;
}

// ===========================================================================
extern "C" void kernel_launch(void* const* d_in, const int* in_sizes, int n_in,
                              void* d_out, int out_size, void* d_ws, size_t ws_size,
                              hipStream_t stream)
{
  (void)in_sizes; (void)n_in; (void)out_size; (void)ws_size;
  const unsigned char* mp_valid = (const unsigned char*)d_in[0];
  const float* mp_attr = (const float*)d_in[1];
  const float* mp_pose = (const float*)d_in[2];
  // d_in[3] = mp_type (unused by reference)
  const float* pe_w1 = (const float*)d_in[4];
  const float* pe_b1 = (const float*)d_in[5];
  const float* pe_w2 = (const float*)d_in[6];
  const float* pe_b2 = (const float*)d_in[7];
  const float* in_wa = (const float*)d_in[8];
  const float* in_ba = (const float*)d_in[9];
  const float* in_wo = (const float*)d_in[10];
  const float* in_bo = (const float*)d_in[11];
  const float* pl_w  = (const float*)d_in[12];
  const float* pl_b  = (const float*)d_in[13];
  const float* pl_wo = (const float*)d_in[14];
  const float* pl_bo = (const float*)d_in[15];
  const float* rpe_w1 = (const float*)d_in[16];
  const float* rpe_b1 = (const float*)d_in[17];
  const float* rpe_w2 = (const float*)d_in[18];
  const float* rpe_b2 = (const float*)d_in[19];
  const float* wq = (const float*)d_in[20];
  const float* bq = (const float*)d_in[21];
  const float* wk = (const float*)d_in[22];
  const float* bk = (const float*)d_in[23];
  const float* wv = (const float*)d_in[24];
  const float* bv = (const float*)d_in[25];
  const float* wo = (const float*)d_in[26];
  const float* bo = (const float*)d_in[27];
  const float* wrpe = (const float*)d_in[28];
  const float* brpe = (const float*)d_in[29];
  const float* ln1g = (const float*)d_in[30];
  const float* ln1b = (const float*)d_in[31];
  const float* ffw1 = (const float*)d_in[32];
  const float* ffb1 = (const float*)d_in[33];
  const float* ffw2 = (const float*)d_in[34];
  const float* ffb2 = (const float*)d_in[35];
  const float* ln2g = (const float*)d_in[36];
  const float* ln2b = (const float*)d_in[37];

  const int M_pts = BB * NTOK * PPTS;   // 61440
  const int M_tok = BB * NTOK;          // 3072
  const int M_rpe = M_tok * KNNK;       // 98304

  // ---- workspace layout (float offsets), regions reused across phases ----
  float* ws = (float*)d_ws;
  float* hbuf   = ws;                        // 61440*256 : h, then hp
  float* h2     = ws + 15728640;             // 61440*256 : h2; later rpe_h/rpe
  float* peh1   = ws + 31457280;             // 61440*128 : pe hidden; later ff1
  float* pein   = ws + 39321600;             // 61440*32 (K padded)
  float* abase  = ws + 41287680;             // 3072*128
  float* pooled = ws + 41680896;             // 3072*256
  float* x      = ws + 42467328;             // 3072*256
  float* q      = ws + 43253760;             // 3072*256
  float* kkbuf  = ws + 44040192;             // 3072*256
  float* vvbuf  = ws + 44826624;             // 3072*256
  float* attb   = ws + 45613056;             // 98304*16 (N padded)
  float* atto   = ws + 47185920;             // 3072*256
  float* resid  = ws + 47972352;             // 3072*256
  float* rpein  = ws + 48758784;             // 98304*32 (K padded)
  int*   knnidx = (int*)(ws + 51904512);     // 98304
  int*   knninv = (int*)(ws + 52002816);     // 98304
  float* rpeh   = h2;                        // reuse (encoder finished)
  float* rpe    = h2 + 6291456;
  float* ff1    = peh1;                      // reuse

  _Float16* wh_base = (_Float16*)(ws + 52101120);
  size_t wh_off = 0;
  auto pack = [&](const float* W, int ldw, int K, int N, int Kpad, int Npad) {
    _Float16* dst = wh_base + wh_off;
    int total = Npad * Kpad;
    pack_w_kernel<<<(total + 255) / 256, 256, 0, stream>>>(W, ldw, K, N, dst, Kpad, total);
    wh_off += (size_t)total;
    return (const _Float16*)dst;
  };

  // ---- pack all weights to f16 [Npad][Kpad] (transposed) -----------------
  const _Float16* pe_w1h  = pack(pe_w1, PE_D, 4, PE_D, 32, PE_D);
  const _Float16* pe_w2h  = pack(pe_w2, PE_D, PE_D, PE_D, PE_D, PE_D);
  const _Float16* in_wah  = pack(in_wa, PE_D, ATTRD, PE_D, ATTRD, PE_D);  // rows 0..31
  const _Float16* in_woh  = pack(in_wo, HH, HH, HH, HH, HH);
  const _Float16* pl_wh   = pack(pl_w,  HH, HH, HH, HH, HH);
  const _Float16* pl_woh  = pack(pl_wo, HH, HH, HH, HH, HH);
  const _Float16* rpe_w1h = pack(rpe_w1, DRPED, 4, DRPED, 32, DRPED);
  const _Float16* rpe_w2h = pack(rpe_w2, DRPED, DRPED, DRPED, DRPED, DRPED);
  const _Float16 *wqh[LLAY], *wkh[LLAY], *wvh[LLAY], *woh[LLAY];
  const _Float16 *wrpeh[LLAY], *ffw1h[LLAY], *ffw2h[LLAY];
  for (int l = 0; l < LLAY; ++l) {
    wqh[l]   = pack(wq + (size_t)l * HH * HH, HH, HH, HH, HH, HH);
    wkh[l]   = pack(wk + (size_t)l * HH * HH, HH, HH, HH, HH, HH);
    wvh[l]   = pack(wv + (size_t)l * HH * HH, HH, HH, HH, HH, HH);
    woh[l]   = pack(wo + (size_t)l * HH * HH, HH, HH, HH, HH, HH);
    wrpeh[l] = pack(wrpe + (size_t)l * DRPED * NHEAD, NHEAD, DRPED, NHEAD, DRPED, 16);
    ffw1h[l] = pack(ffw1 + (size_t)l * HH * DFFD, DFFD, HH, DFFD, HH, DFFD);
    ffw2h[l] = pack(ffw2 + (size_t)l * DFFD * HH, HH, DFFD, HH, DFFD, HH);
  }

  auto gemm = [&](const float* A, int lda, const _Float16* Wh,
                  const float* bias, int biasN, float* C, int ldc,
                  int M, int Kpad, int Npad, int act) {
    if (Npad % 64 == 0) {
      long waves = (long)(M / 16) * (Npad / 64);
      int blocks = (int)((waves + 7) / 8);
      gemm_wmma_f16<4><<<blocks, 256, 0, stream>>>(A, lda, Wh, bias, biasN, C, ldc,
                                                   M, Kpad, Npad, act);
    } else {
      long waves = (long)(M / 16) * (Npad / 16);
      int blocks = (int)((waves + 7) / 8);
      gemm_wmma_f16<1><<<blocks, 256, 0, stream>>>(A, lda, Wh, bias, biasN, C, ldc,
                                                   M, Kpad, Npad, act);
    }
  };

  // ---- encoder ----
  prep_points_kernel<<<(M_pts + 255) / 256, 256, 0, stream>>>(mp_pose, pein, M_pts);
  gemm(mp_attr, ATTRD, in_wah, nullptr, 0, abase, 128, M_tok, ATTRD, 128, 0);
  gemm(pein, 32, pe_w1h, pe_b1, PE_D, peh1, PE_D, M_pts, 32, PE_D, 1);
  gemm(peh1, PE_D, pe_w2h, pe_b2, PE_D, hbuf + 128, HH, M_pts, PE_D, PE_D, 0);
  ha_kernel<<<(M_pts * 128 + 255) / 256, 256, 0, stream>>>(abase, in_wa, in_ba, hbuf, M_pts * 128);
  gemm(hbuf, HH, in_woh, in_bo, HH, h2, HH, M_pts, HH, HH, 1);
  gemm(h2, HH, pl_wh, pl_b, HH, hbuf, HH, M_pts, HH, HH, 1);    // hp reuses hbuf
  pool_kernel<<<M_tok, 256, 0, stream>>>(hbuf, mp_valid, pooled);
  gemm(pooled, HH, pl_woh, pl_bo, HH, x, HH, M_tok, HH, HH, 0);

  // ---- KNN graph + relative-pose encoding ----
  knn_kernel<<<M_tok, 256, 0, stream>>>(mp_pose, mp_valid, knnidx, knninv, rpein);
  gemm(rpein, 32, rpe_w1h, rpe_b1, DRPED, rpeh, DRPED, M_rpe, 32, DRPED, 1);
  gemm(rpeh, DRPED, rpe_w2h, rpe_b2, DRPED, rpe, DRPED, M_rpe, DRPED, DRPED, 0);

  // ---- transformer layers ----
  for (int l = 0; l < LLAY; ++l) {
    gemm(x, HH, wqh[l], bq + (size_t)l * HH, HH, q, HH, M_tok, HH, HH, 0);
    gemm(x, HH, wkh[l], bk + (size_t)l * HH, HH, kkbuf, HH, M_tok, HH, HH, 0);
    gemm(x, HH, wvh[l], bv + (size_t)l * HH, HH, vvbuf, HH, M_tok, HH, HH, 0);
    gemm(rpe, DRPED, wrpeh[l], brpe + (size_t)l * NHEAD, NHEAD, attb, 16,
         M_rpe, DRPED, 16, 0);
    attn_kernel<<<M_tok, 256, 0, stream>>>(q, kkbuf, vvbuf, attb, knnidx, knninv, atto);
    gemm(atto, HH, woh[l], bo + (size_t)l * HH, HH, resid, HH, M_tok, HH, HH, 0);
    add_ln_kernel<<<M_tok, 256, 0, stream>>>(x, resid,
        ln1g + (size_t)l * HH, ln1b + (size_t)l * HH);
    gemm(x, HH, ffw1h[l], ffb1 + (size_t)l * DFFD, DFFD, ff1, DFFD, M_tok, HH, DFFD, 1);
    gemm(ff1, DFFD, ffw2h[l], ffb2 + (size_t)l * HH, HH, resid, HH, M_tok, DFFD, HH, 0);
    add_ln_kernel<<<M_tok, 256, 0, stream>>>(x, resid,
        ln2g + (size_t)l * HH, ln2b + (size_t)l * HH);
  }

  final_kernel<<<(M_tok * HH + 255) / 256, 256, 0, stream>>>(x, mp_valid, (float*)d_out);
}
